// CPN_11845519802814
// MI455X (gfx1250) — matrix operand order repaired and checked by reading, not compiled, not run on hardware
//
#include <hip/hip_runtime.h>
#include <hip/hip_bf16.h>
#include <math.h>

typedef __attribute__((ext_vector_type(16))) _Float16 v16h;
typedef __attribute__((ext_vector_type(8)))  _Float16 v8h;
typedef __attribute__((ext_vector_type(8)))  float    v8f;
typedef __attribute__((ext_vector_type(4)))  unsigned int u32x4;
typedef __attribute__((ext_vector_type(8)))  int          i32x8;
typedef __attribute__((ext_vector_type(4)))  int          i32x4;
// async-load builtin wants typed, address-space-qualified pointers:
typedef __attribute__((address_space(1))) i32x4 g_i32x4;   // global int4
typedef __attribute__((address_space(3))) i32x4 l_i32x4;   // LDS int4

#define HH 768
#define WW 768
#define NB 2
#define CBK 32   // backbone / head mid channels

// --- CDNA5 feature detection (compile-safe fallbacks) -----------------------
#if defined(__has_builtin)
#  if __has_builtin(__builtin_amdgcn_tensor_load_to_lds) && \
      __has_builtin(__builtin_amdgcn_s_wait_tensorcnt)
#    define HAS_TDM 1
#  endif
#  if __has_builtin(__builtin_amdgcn_global_load_async_to_lds_b128) && \
      __has_builtin(__builtin_amdgcn_s_wait_asynccnt)
#    define HAS_ASYNC 1
#  endif
#endif
#if __has_include(<hip/amd_detail/amd_gfx1250_TDM.h>)
#  define TDM_SIX_ARGS 1   // therock-10.0 headers => 6-arg builtin
#endif

// ---------------------------------------------------------------------------
// Repack head conv1 weights [O=32][I=32][KW][KW] f32  ->  per-tap WMMA B tiles
// layout: rep[(tap*2 + coTile)*32 + lane][16] f16, per ISA 7.12.2 B layout:
//   lanes 0-15  hold K rows 0..7,16..23 (2/VGPR), N(col)=lane
//   lanes 16-31 hold K rows 8..15,24..31,         N(col)=lane-16
// ---------------------------------------------------------------------------
__global__ void repack_kernel(const float* __restrict__ w1,
                              _Float16* __restrict__ rep, int KW)
{
    const int taps = KW * KW;
    const int idx  = blockIdx.x * 64 + threadIdx.x;   // taps*2*32 threads
    if (idx >= taps * 64) return;
    const int lane  = idx & 31;
    const int ct    = (idx >> 5) & 1;
    const int tap   = idx >> 6;
    const int o     = ct * 16 + (lane & 15);
    const int ibase = (lane >> 4) ? 16 : 0;
    const int dy    = tap / KW, dx = tap % KW;
    _Float16* dst = rep + (size_t)idx * 16;
    #pragma unroll 4
    for (int j = 0; j < 16; ++j)
        dst[j] = (_Float16)w1[(((size_t)o * CBK + (ibase / 2 + j + (j & 8))) * KW + dy) * KW + dx];
}

// ---------------------------------------------------------------------------
// Backbone: 3x3 conv 3->32 + ReLU, NCHW f32 in -> NHWC f16 out
// ---------------------------------------------------------------------------
__global__ __launch_bounds__(256) void backbone_kernel(
    const float* __restrict__ in, const float* __restrict__ w,
    const float* __restrict__ bias, _Float16* __restrict__ feats)
{
    const size_t idx = (size_t)blockIdx.x * 256 + threadIdx.x;   // 2*768*768 exact
    const int x = (int)(idx % WW);
    const int y = (int)((idx / WW) % HH);
    const int b = (int)(idx / ((size_t)HH * WW));

    float vin[3][3][3];
    #pragma unroll
    for (int ci = 0; ci < 3; ++ci)
        #pragma unroll
        for (int dy = 0; dy < 3; ++dy)
            #pragma unroll
            for (int dx = 0; dx < 3; ++dx) {
                const int gy = y + dy - 1, gx = x + dx - 1;
                vin[ci][dy][dx] = (gy >= 0 && gy < HH && gx >= 0 && gx < WW)
                    ? in[(((size_t)b * 3 + ci) * HH + gy) * WW + gx] : 0.f;
            }

    __align__(16) _Float16 o[CBK];
    for (int co = 0; co < CBK; ++co) {
        float a = bias[co];
        #pragma unroll
        for (int ci = 0; ci < 3; ++ci)
            #pragma unroll
            for (int dy = 0; dy < 3; ++dy)
                #pragma unroll
                for (int dx = 0; dx < 3; ++dx)
                    a = fmaf(vin[ci][dy][dx], w[((co * 3 + ci) * 3 + dy) * 3 + dx], a);
        o[co] = (_Float16)(a > 0.f ? a : 0.f);
    }
    uint4* dst = (uint4*)(feats + (((size_t)b * HH + y) * WW + x) * CBK);
    const uint4* src = (const uint4*)o;
    dst[0] = src[0]; dst[1] = src[1]; dst[2] = src[2]; dst[3] = src[3];
}

// ---------------------------------------------------------------------------
// Fused head: KWxKW conv 32->32 (one 16x16x32 WMMA per tap per cout-tile),
// folded-BN + ReLU, fused 1x1 conv 32->cout2 (+ optional tanh*3).
// Block = 8 waves; wave w owns row y0+w, 16-pixel x strip.
// Patch staging: TDM tensor_load_to_lds (interior blocks) or
// global_load_async_to_lds_b128 / vector-copy fallback (border blocks).
// ---------------------------------------------------------------------------
template<int KW, bool DO_TANH>
__global__ __launch_bounds__(256) void head_kernel(
    const _Float16* __restrict__ feats,
    const _Float16* __restrict__ repw1,
    const float* __restrict__ sBN, const float* __restrict__ b1,
    const float* __restrict__ tBN,
    const float* __restrict__ w2, const float* __restrict__ b2,
    float* __restrict__ out, int cout2)
{
    constexpr int P  = KW / 2;
    constexpr int PR = 8 + 2 * P;
    constexpr int PC = 16 + 2 * P;

    __shared__ __align__(64) _Float16 patch[PR][PC][CBK];  // halo tile, NHWC
    __shared__ __align__(64) _Float16 hsm[8][16][CBK];     // post-ReLU h tile

    const int tid = threadIdx.x;
    const int x0  = blockIdx.x * 16;
    const int y0  = blockIdx.y * 8;
    const int b   = blockIdx.z;

#if HAS_TDM
    const bool interior = (x0 >= P) && (x0 + 16 + P <= WW) &&
                          (y0 >= P) && (y0 + 8 + P <= HH);
    if (interior) {
        if (tid < 32) {
            // --- build D# for a 2D tile of the NHWC f16 feats tensor ---
            const unsigned long long gaddr = (unsigned long long)
                (feats + (((size_t)b * HH + (y0 - P)) * WW + (x0 - P)) * CBK);
            const unsigned ldsa = (unsigned)(unsigned long long)&patch[0][0][0];
            const unsigned td0  = (unsigned)(PC * CBK);   // elements per tile row
            const unsigned td1  = (unsigned)PR;           // rows
            const unsigned str0 = (unsigned)(WW * CBK);   // row stride (elements)
            u32x4 g0;
            g0[0] = 1u;                                   // count=1, user mode
            g0[1] = ldsa;                                 // lds_addr (bytes)
            g0[2] = (unsigned)(gaddr & 0xffffffffu);      // global_addr[31:0]
            g0[3] = (unsigned)((gaddr >> 32) & 0x01ffffffu) | (2u << 30); // [56:32] | type=2
            i32x8 g1;
            g1[0] = (int)(1u << 16);                      // data_size=1 (2B), mask=0
            g1[1] = (int)((td0 & 0xffffu) << 16);         // tensor_dim0[15:0] @ bit48
            g1[2] = (int)((td0 >> 16) | ((td1 & 0xffffu) << 16)); // dim0 hi | dim1 lo
            g1[3] = (int)((td1 >> 16) | (td0 << 16));     // dim1 hi | tile_dim0
            g1[4] = (int)(td1 & 0xffffu);                 // tile_dim1 | tile_dim2=0
            g1[5] = (int)str0;                            // tensor_dim0_stride[31:0]
            g1[6] = 0;                                    // stride0 hi | stride1 lo
            g1[7] = 0;
            i32x4 z4; z4[0] = 0; z4[1] = 0; z4[2] = 0; z4[3] = 0;
#if TDM_SIX_ARGS
            i32x8 z8; for (int i = 0; i < 8; ++i) z8[i] = 0;
            __builtin_amdgcn_tensor_load_to_lds(g0, g1, z4, z4, z8, 0);
#else
            __builtin_amdgcn_tensor_load_to_lds(g0, g1, z4, z4, 0);
#endif
            __builtin_amdgcn_s_wait_tensorcnt(0);
        }
    } else
#endif
    {
        for (int idx = tid; idx < PR * PC; idx += 256) {
            const int row = idx / PC, col = idx % PC;
            const int gy = y0 - P + row, gx = x0 - P + col;
            _Float16* lp = &patch[row][col][0];
            if (gy >= 0 && gy < HH && gx >= 0 && gx < WW) {
                const _Float16* gp =
                    feats + (((size_t)b * HH + gy) * WW + gx) * CBK;
#if HAS_ASYNC
                g_i32x4* g = (g_i32x4*)(unsigned long long)gp;
                l_i32x4* l = (l_i32x4*)(unsigned long long)lp;
                __builtin_amdgcn_global_load_async_to_lds_b128(g, l, 0,  0);
                __builtin_amdgcn_global_load_async_to_lds_b128(g, l, 16, 0);
                __builtin_amdgcn_global_load_async_to_lds_b128(g, l, 32, 0);
                __builtin_amdgcn_global_load_async_to_lds_b128(g, l, 48, 0);
#else
                const uint4* src = (const uint4*)gp;
                uint4* dst = (uint4*)lp;
                dst[0] = src[0]; dst[1] = src[1]; dst[2] = src[2]; dst[3] = src[3];
#endif
            } else {
                const uint4 z = {0, 0, 0, 0};
                uint4* dst = (uint4*)lp;
                dst[0] = z; dst[1] = z; dst[2] = z; dst[3] = z;
            }
        }
#if HAS_ASYNC
        __builtin_amdgcn_s_wait_asynccnt(0);
#endif
    }
    __syncthreads();

    const int lane = tid & 31;
    const int wave = tid >> 5;
    const int m    = lane & 15;        // pixel in strip == A-matrix M lane slot
    const int hiL  = lane >> 4;
    const int koff = hiL ? 8 : 0;      // A-matrix K base for this lane (ISA 7.12.2)

    v8f acc0 = {};                     // cout 0..15
    v8f acc1 = {};                     // cout 16..31

    for (int dy = 0; dy < KW; ++dy) {
        #pragma unroll
        for (int dx = 0; dx < KW; ++dx) {
            const _Float16* ap = &patch[wave + dy][m + dx][0];
            const v8h alo = *(const v8h*)(ap + koff);
            const v8h ahi = *(const v8h*)(ap + koff + 16);
            v16h A;
            #pragma unroll
            for (int i = 0; i < 8; ++i) { A[i] = alo[i]; A[8 + i] = ahi[i]; }

            const int tap = dy * KW + dx;
            const v16h B0 = *(const v16h*)(repw1 + ((size_t)(tap * 2 + 0) * 32 + lane) * 16);
            const v16h B1 = *(const v16h*)(repw1 + ((size_t)(tap * 2 + 1) * 32 + lane) * 16);

            acc0 = __builtin_amdgcn_wmma_f32_16x16x32_f16(
                       false, A, false, B0, (short)0, acc0, false, false);
            acc1 = __builtin_amdgcn_wmma_f32_16x16x32_f16(
                       false, A, false, B1, (short)0, acc1, false, false);
        }
    }

    // ---- folded BN + ReLU; spill h tile to LDS (lane's channel = m) ----
    const int ch0 = m, ch1 = m + 16;
    const float sc0 = sBN[ch0], sh0 = b1[ch0] * sc0 + tBN[ch0];
    const float sc1 = sBN[ch1], sh1 = b1[ch1] * sc1 + tBN[ch1];
    #pragma unroll
    for (int r = 0; r < 8; ++r) {
        const int px = hiL ? (8 + r) : r;
        float u0 = acc0[r] * sc0 + sh0; u0 = u0 > 0.f ? u0 : 0.f;
        float u1 = acc1[r] * sc1 + sh1; u1 = u1 > 0.f ? u1 : 0.f;
        hsm[wave][px][ch0] = (_Float16)u0;
        hsm[wave][px][ch1] = (_Float16)u1;
    }
    __syncthreads();

    // ---- fused 1x1 conv 32 -> cout2 (+ bias, optional tanh*3), NCHW f32 ----
    const int y = y0 + wave, x = x0 + m;
    const int half  = cout2 >> 1;
    const int coBeg = hiL ? half : 0;
    const int coEnd = hiL ? cout2 : half;
    const _Float16* hp = &hsm[wave][m][0];
    for (int co = coBeg; co < coEnd; ++co) {
        float a = b2[co];
        #pragma unroll
        for (int c = 0; c < CBK; ++c) a += (float)hp[c] * w2[co * CBK + c];
        if (DO_TANH) a = tanhf(a) * 3.0f;
        out[(((size_t)b * cout2 + co) * HH + y) * WW + x] = a;
    }
}

// ---------------------------------------------------------------------------
extern "C" void kernel_launch(void* const* d_in, const int* in_sizes, int n_in,
                              void* d_out, int out_size, void* d_ws, size_t ws_size,
                              hipStream_t stream)
{
    (void)in_sizes; (void)n_in; (void)out_size; (void)ws_size;

    const float* inputs = (const float*)d_in[0];
    const float* bb_w   = (const float*)d_in[1];
    const float* bb_b   = (const float*)d_in[2];
    // heads in dict order: 0=score, 1=loc, 2=four, 3=ref; fields: w1,b1,s,t,w2,b2
    auto W1 = [&](int h) { return (const float*)d_in[3 + 6 * h + 0]; };
    auto B1 = [&](int h) { return (const float*)d_in[3 + 6 * h + 1]; };
    auto SS = [&](int h) { return (const float*)d_in[3 + 6 * h + 2]; };
    auto TT = [&](int h) { return (const float*)d_in[3 + 6 * h + 3]; };
    auto W2 = [&](int h) { return (const float*)d_in[3 + 6 * h + 4]; };
    auto B2 = [&](int h) { return (const float*)d_in[3 + 6 * h + 5]; };

    char* ws = (char*)d_ws;
    size_t off = 0;
    auto alloc = [&](size_t bytes) -> char* {
        char* p = ws + off;
        off = (off + bytes + 255) & ~(size_t)255;
        return p;
    };
    _Float16* feats = (_Float16*)alloc((size_t)NB * HH * WW * CBK * sizeof(_Float16));
    _Float16* repS  = (_Float16*)alloc((size_t)9  * 64 * 16 * sizeof(_Float16));
    _Float16* repL  = (_Float16*)alloc((size_t)49 * 64 * 16 * sizeof(_Float16));
    _Float16* repF  = (_Float16*)alloc((size_t)49 * 64 * 16 * sizeof(_Float16));
    _Float16* repR  = (_Float16*)alloc((size_t)49 * 64 * 16 * sizeof(_Float16));

    repack_kernel<<<9,  64, 0, stream>>>(W1(0), repS, 3);
    repack_kernel<<<49, 64, 0, stream>>>(W1(1), repL, 7);
    repack_kernel<<<49, 64, 0, stream>>>(W1(2), repF, 7);
    repack_kernel<<<49, 64, 0, stream>>>(W1(3), repR, 7);

    backbone_kernel<<<(NB * HH * WW) / 256, 256, 0, stream>>>(inputs, bb_w, bb_b, feats);

    float* out = (float*)d_out;
    const size_t plane2 = (size_t)NB * 2 * HH * WW;   // 2,359,296 floats
    dim3 grid(WW / 16, HH / 8, NB), block(256);
    head_kernel<3, false><<<grid, block, 0, stream>>>(feats, repS, SS(0), B1(0), TT(0),
                                                      W2(0), B2(0), out, 2);
    head_kernel<7, false><<<grid, block, 0, stream>>>(feats, repL, SS(1), B1(1), TT(1),
                                                      W2(1), B2(1), out + plane2, 2);
    head_kernel<7, true ><<<grid, block, 0, stream>>>(feats, repR, SS(3), B1(3), TT(3),
                                                      W2(3), B2(3), out + 2 * plane2, 2);
    head_kernel<7, false><<<grid, block, 0, stream>>>(feats, repF, SS(2), B1(2), TT(2),
                                                      W2(2), B2(2), out + 3 * plane2, 20);
}
